// DWALanguageModel_77498389889527
// MI455X (gfx1250) — matrix operand order repaired and checked by reading, not compiled, not run on hardware
//
#include <hip/hip_runtime.h>
#include <hip/hip_bf16.h>
#include <math.h>

// ---------------------------------------------------------------------------
// Problem constants (from the reference)
// ---------------------------------------------------------------------------
#define BB      1024      // batch
#define NPOOL   1024      // pool entries
#define DPOOL   16384     // pool dim
#define DMODEL  512
#define RRANK   8
#define SS      2
#define DKQ     64
#define KMAX    16
#define QKDIM   (SS * DKQ)   // 128 (both aspects concatenated)

typedef __attribute__((ext_vector_type(16))) __bf16        bf16x16;
typedef __attribute__((ext_vector_type(8)))  float         v8f;
typedef __attribute__((ext_vector_type(4)))  unsigned int  u32x4;
typedef __attribute__((ext_vector_type(8)))  int           i32x8;
typedef __attribute__((ext_vector_type(4)))  int           i32x4;

// ---------------------------------------------------------------------------
// WMMA fragment loaders (CDNA5 wave32 layouts, ISA 7.12.2)
//  A (16x32, 16-bit): lanes 0-15 -> row=lane, K = {0..7, 16..23}
//                     lanes 16-31 -> row=lane-16, K = {8..15, 24..31}
//  B via transposed source Bt[col][k]: lanes 0-15 -> col=lane, K = {0..7, 16..23}
//                                      lanes 16-31 -> col=lane-16, K = {8..15, 24..31}
// ---------------------------------------------------------------------------
template <typename T>
__device__ inline bf16x16 frag_A(const T* __restrict__ A, int ld, int row, int k0) {
  int lane = threadIdx.x & 31;
  const T* p = A + (size_t)row * ld + k0 + ((lane & 16) ? 8 : 0);
  bf16x16 f;
#pragma unroll
  for (int i = 0; i < 8; ++i) f[i] = (__bf16)p[i];
#pragma unroll
  for (int i = 0; i < 8; ++i) f[8 + i] = (__bf16)p[16 + i];
  return f;
}

template <typename T>
__device__ inline bf16x16 frag_BT(const T* __restrict__ Bt, int ld, int col, int k0) {
  int lane = threadIdx.x & 31;
  const T* p = Bt + (size_t)col * ld + k0 + ((lane & 16) ? 8 : 0);
  bf16x16 f;
#pragma unroll
  for (int i = 0; i < 8; ++i) f[i] = (__bf16)p[i];
#pragma unroll
  for (int i = 0; i < 8; ++i) f[8 + i] = (__bf16)p[16 + i];
  return f;
}

// ---------------------------------------------------------------------------
// Tensor Data Mover: 2-D tile load (f32 elements) into LDS.
// D# per ISA ch.8: group0 = {count/flags, lds_addr, global_addr, type=2},
// group1 = {mask/data_size, tensor dims, tile dims, dim0 stride}.
// Issued by one wave; completion fenced with s_wait_tensorcnt.
// ---------------------------------------------------------------------------
__device__ inline unsigned lds_addr_of(const void* p) {
  return (unsigned)(unsigned long long)(uintptr_t)p;  // low 32 bits = LDS offset
}

__device__ inline void tdm_load_2d_f32(unsigned lds_addr, const float* gptr,
                                       unsigned tile_x, unsigned tile_y,
                                       unsigned tensor_x, unsigned tensor_y,
                                       unsigned stride_elems) {
  unsigned long long ga = (unsigned long long)(uintptr_t)gptr;
  u32x4 g0;
  g0[0] = 1u;                                   // count=1 (valid), user mode
  g0[1] = lds_addr;                             // lds_addr [63:32]
  g0[2] = (unsigned)(ga & 0xFFFFFFFFu);         // global_addr[31:0]
  g0[3] = (unsigned)((ga >> 32) & 0x1FFFFFFu)   // global_addr[56:32]
        | (2u << 30);                           // type = 2 ("image")
  i32x8 g1;
  g1[0] = (int)(2u << 16);                      // wg_mask=0, data_size=2 (4 B)
  g1[1] = (int)((tensor_x & 0xFFFFu) << 16);    // tensor_dim0[15:0]
  g1[2] = (int)(((tensor_x >> 16) & 0xFFFFu) | ((tensor_y & 0xFFFFu) << 16));
  g1[3] = (int)(((tensor_y >> 16) & 0xFFFFu) | ((tile_x & 0xFFFFu) << 16));
  g1[4] = (int)(tile_y & 0xFFFFu);              // tile_dim1; tile_dim2 = 0
  g1[5] = (int)stride_elems;                    // tensor_dim0_stride[31:0]
  g1[6] = 0;                                    // stride[47:32]=0, dim1_stride lo
  g1[7] = 0;
  i32x4 z4 = {0, 0, 0, 0};                      // 2-D: groups 2/3 unused
  i32x8 z8 = {0, 0, 0, 0, 0, 0, 0, 0};
  __builtin_amdgcn_tensor_load_to_lds(g0, g1, z4, z4, z8, 0);
}

// ---------------------------------------------------------------------------
// keys = pool @ W_K^T-prepped. M=1024, N=128, K=16384.
// A-tile (64 rows x 32 k, f32) streamed by TDM into double-buffered LDS;
// B fragments register-double-buffered from the L2-resident bf16 W_K^T.
// Block = 128 threads (4 waves); block tile 64(M) x 64(N).
// ---------------------------------------------------------------------------
__global__ void wmma_gemm_pool_tdm(const float* __restrict__ A,
                                   const __bf16* __restrict__ Bt,
                                   float* __restrict__ C) {
  __shared__ float abuf[2][64 * 32];   // 16 KB double buffer
  const int wave = threadIdx.x >> 5;
  const int lane = threadIdx.x & 31;
  const int m0 = blockIdx.y * 64;
  const int n0 = blockIdx.x * 64;
  const int lrow = wave * 16 + (lane & 15);

  // prologue: DMA first A tile, preload first B fragments
  if (threadIdx.x < 32) {
    tdm_load_2d_f32(lds_addr_of(&abuf[0][0]), A + (size_t)m0 * DPOOL,
                    32, 64, DPOOL, NPOOL, DPOOL);
    __builtin_amdgcn_s_wait_tensorcnt(0);
  }
  bf16x16 b[4];
#pragma unroll
  for (int nt = 0; nt < 4; ++nt)
    b[nt] = frag_BT(Bt, DPOOL, n0 + nt * 16 + (lane & 15), 0);
  __syncthreads();

  v8f acc[4] = {v8f{}, v8f{}, v8f{}, v8f{}};

  for (int k0 = 0; k0 < DPOOL; k0 += 32) {
    const int cur = (k0 >> 5) & 1;
    const bool more = (k0 + 32) < DPOOL;

    // kick off DMA of the next A tile while we compute on the current one
    if (threadIdx.x < 32 && more) {
      tdm_load_2d_f32(lds_addr_of(&abuf[cur ^ 1][0]),
                      A + (size_t)m0 * DPOOL + (k0 + 32),
                      32, 64, DPOOL, NPOOL, DPOOL);
    }
    // prefetch next B fragments into registers
    bf16x16 bn[4] = {b[0], b[1], b[2], b[3]};
    if (more) {
#pragma unroll
      for (int nt = 0; nt < 4; ++nt)
        bn[nt] = frag_BT(Bt, DPOOL, n0 + nt * 16 + (lane & 15), k0 + 32);
    }

    // A fragment from the current LDS tile (f32 -> bf16)
    bf16x16 a;
    {
      const float* p = &abuf[cur][lrow * 32 + ((lane & 16) ? 8 : 0)];
#pragma unroll
      for (int i = 0; i < 8; ++i) a[i] = (__bf16)p[i];
#pragma unroll
      for (int i = 0; i < 8; ++i) a[8 + i] = (__bf16)p[16 + i];
    }

#pragma unroll
    for (int nt = 0; nt < 4; ++nt)
      acc[nt] = __builtin_amdgcn_wmma_f32_16x16x32_bf16(
          false, a, false, b[nt], (short)0, acc[nt], false, false);

#pragma unroll
    for (int nt = 0; nt < 4; ++nt) b[nt] = bn[nt];

    __syncthreads();                              // everyone done with abuf[cur]
    if (threadIdx.x < 32 && more)
      __builtin_amdgcn_s_wait_tensorcnt(0);       // next tile landed
    __syncthreads();
  }

  const int rbase = m0 + wave * 16 + ((lane & 16) ? 8 : 0);
#pragma unroll
  for (int nt = 0; nt < 4; ++nt) {
    const int col = n0 + nt * 16 + (lane & 15);
#pragma unroll
    for (int i = 0; i < 8; ++i)
      C[(size_t)(rbase + i) * QKDIM + col] = acc[nt][i];
  }
}

// ---------------------------------------------------------------------------
// Generic C = A (MxK row-major) * Bt^T (Bt NxK row-major), f32 accumulate,
// register double-buffered so next-step loads overlap current WMMAs.
// Block = 128 threads (4 waves); block tile 64(M) x 64(N).
// ---------------------------------------------------------------------------
template <typename TA>
__global__ void wmma_gemm_nt(const TA* __restrict__ A, const __bf16* __restrict__ Bt,
                             float* __restrict__ C, int M, int N, int K,
                             int lda, int ldb, int ldc) {
  const int wave = threadIdx.x >> 5;
  const int lane = threadIdx.x & 31;
  const int m0 = blockIdx.y * 64 + wave * 16;
  const int n0 = blockIdx.x * 64;
  if (m0 >= M || n0 >= N) return;

  v8f acc[4] = {v8f{}, v8f{}, v8f{}, v8f{}};
  const int arow = m0 + (lane & 15);

  bf16x16 a = frag_A(A, lda, arow, 0);
  bf16x16 b[4];
#pragma unroll
  for (int nt = 0; nt < 4; ++nt)
    b[nt] = frag_BT(Bt, ldb, n0 + nt * 16 + (lane & 15), 0);

  for (int k0 = 0; k0 < K; k0 += 32) {
    bf16x16 an = a;
    bf16x16 bn[4] = {b[0], b[1], b[2], b[3]};
    if (k0 + 32 < K) {
      an = frag_A(A, lda, arow, k0 + 32);
#pragma unroll
      for (int nt = 0; nt < 4; ++nt)
        bn[nt] = frag_BT(Bt, ldb, n0 + nt * 16 + (lane & 15), k0 + 32);
    }
#pragma unroll
    for (int nt = 0; nt < 4; ++nt)
      acc[nt] = __builtin_amdgcn_wmma_f32_16x16x32_bf16(
          false, a, false, b[nt], (short)0, acc[nt], false, false);
    a = an;
#pragma unroll
    for (int nt = 0; nt < 4; ++nt) b[nt] = bn[nt];
  }

  const int rbase = m0 + ((lane & 16) ? 8 : 0);
#pragma unroll
  for (int nt = 0; nt < 4; ++nt) {
    const int col = n0 + nt * 16 + (lane & 15);
#pragma unroll
    for (int i = 0; i < 8; ++i)
      C[(size_t)(rbase + i) * ldc + col] = acc[nt][i];
  }
}

// ---------------------------------------------------------------------------
// W_K/W_Q (S, Kdim, 64) f32  ->  Bt (128, Kdim) bf16   (col = s*64 + q)
// ---------------------------------------------------------------------------
__global__ void transpose_w_bf16(const float* __restrict__ W, __bf16* __restrict__ Bt,
                                 int Kdim, int total) {
  int i = blockIdx.x * blockDim.x + threadIdx.x;
  if (i >= total) return;
  int q = i & 63;
  int k = (i >> 6) % Kdim;
  int s = (i >> 6) / Kdim;
  Bt[(size_t)(s * 64 + q) * Kdim + k] = (__bf16)W[i];
}

__global__ void convert_bf16(const float* __restrict__ X, __bf16* __restrict__ Y, int total) {
  int i = blockIdx.x * blockDim.x + threadIdx.x;
  if (i < total) Y[i] = (__bf16)X[i];
}

// ---------------------------------------------------------------------------
// Per-row L2 normalization of the two 64-wide aspect segments, scaled by
// sqrt(softmax(aspect_logits)_s), emitted as bf16 for the scores GEMM.
// ---------------------------------------------------------------------------
__global__ void normalize_qk(const float* __restrict__ X, __bf16* __restrict__ Y,
                             const float* __restrict__ logits) {
  const int b = blockIdx.x;
  const int t = threadIdx.x;  // 0..127
  __shared__ float sq[QKDIM];
  __shared__ float scale[SS];
  float v = X[(size_t)b * QKDIM + t];
  sq[t] = v * v;
  __syncthreads();
  if (t < SS) {
    float s = 0.f;
    for (int i = 0; i < DKQ; ++i) s += sq[t * DKQ + i];
    float e0 = __expf(logits[0]), e1 = __expf(logits[1]);
    float w = ((t == 0) ? e0 : e1) / (e0 + e1);
    scale[t] = sqrtf(w) / (sqrtf(s) + 1e-8f);
  }
  __syncthreads();
  Y[(size_t)b * QKDIM + t] = (__bf16)(v * scale[t >> 6]);
}

// ---------------------------------------------------------------------------
// Top-16 per row of scores (1024 wide) + sparse alpha. One wave per row.
// ---------------------------------------------------------------------------
__global__ void topk_alpha(const float* __restrict__ scores, const float* __restrict__ tau_p,
                           float* __restrict__ alpha, int* __restrict__ idx_out) {
  const int b = blockIdx.x;
  const int l = threadIdx.x;  // 0..31
  __shared__ float sh[NPOOL];
  __shared__ float selv[KMAX];
  __shared__ int   seli[KMAX];
  for (int j = l; j < NPOOL; j += 32) sh[j] = scores[(size_t)b * NPOOL + j];
  __syncthreads();

  for (int it = 0; it < KMAX; ++it) {
    float best = -1e30f;
    int   bi   = -1;
    for (int j = l; j < NPOOL; j += 32) {
      float v = sh[j];
      if (v > best) { best = v; bi = j; }
    }
#pragma unroll
    for (int off = 16; off > 0; off >>= 1) {
      float ov = __shfl_xor(best, off, 32);
      int   oi = __shfl_xor(bi, off, 32);
      if (ov > best || (ov == best && oi < bi)) { best = ov; bi = oi; }
    }
    if (l == 0) { selv[it] = best; seli[it] = bi; sh[bi] = -1e30f; }
    __syncthreads();
  }

  const float tau = *tau_p;
  float raw = 0.f;
  if (l < KMAX) {
    float v = selv[l];
    float g = 1.0f / (1.0f + __expf(-5.0f * (v - tau)));
    raw = g * __expf(v);  // TEMP = 1
  }
  float tot = raw;
#pragma unroll
  for (int off = 16; off > 0; off >>= 1) tot += __shfl_xor(tot, off, 32);
  if (l < KMAX) {
    alpha[(size_t)b * KMAX + l] = raw / (tot + 1e-8f);
    idx_out[(size_t)b * KMAX + l] = seli[l];
  }
}

// ---------------------------------------------------------------------------
// Per batch-row b:  h = h_base + sum_n alpha * ( U_n (V_n z) + bias_n ),
// then out = LayerNorm(z + gamma * (h + b_base)). One block (256 thr) per row.
// Pool rows are gathered; whole pool (64 MB) is L2-resident (192 MB L2).
// ---------------------------------------------------------------------------
__global__ void assemble_ln(const float* __restrict__ z, const float* __restrict__ pool,
                            const float* __restrict__ h_base, const float* __restrict__ alpha,
                            const int* __restrict__ idxs, const float* __restrict__ b_base,
                            const float* __restrict__ gamma_p, const float* __restrict__ ln_scale,
                            const float* __restrict__ ln_bias, float* __restrict__ out) {
  const int b = blockIdx.x;
  const int t = threadIdx.x;  // 0..255
  __shared__ float zb[DMODEL];
  __shared__ float tv[RRANK];
  __shared__ float red[256];

  zb[t]       = z[(size_t)b * DMODEL + t];
  zb[t + 256] = z[(size_t)b * DMODEL + t + 256];
  float acc0 = h_base[(size_t)b * DMODEL + t];
  float acc1 = h_base[(size_t)b * DMODEL + t + 256];
  __syncthreads();

  const int w = t >> 5, l = t & 31;
  for (int i = 0; i < KMAX; ++i) {
    const int   n = idxs[(size_t)b * KMAX + i];
    const float a = alpha[(size_t)b * KMAX + i];
    const float* pn = pool + (size_t)n * DPOOL;
    // t[r] = V_n[r,:] . z_b   (warp w computes r = w)
    const float* Vr = pn + DMODEL * RRANK + w * DMODEL;  // V at offset 4096
    float s = 0.f;
    for (int j = l; j < DMODEL; j += 32) s += Vr[j] * zb[j];
#pragma unroll
    for (int off = 16; off > 0; off >>= 1) s += __shfl_xor(s, off, 32);
    if (l == 0) tv[w] = s;
    __syncthreads();
    // h_c += a * ( U_n[c,:] . t + bias_n[c] )
    float u0 = 0.f, u1 = 0.f;
#pragma unroll
    for (int r = 0; r < RRANK; ++r) {
      u0 += pn[(size_t)t * RRANK + r] * tv[r];
      u1 += pn[(size_t)(t + 256) * RRANK + r] * tv[r];
    }
    acc0 += a * (u0 + pn[2 * DMODEL * RRANK + t]);
    acc1 += a * (u1 + pn[2 * DMODEL * RRANK + t + 256]);
    __syncthreads();
  }

  const float gamma = *gamma_p;
  float x0 = zb[t]       + gamma * (acc0 + b_base[t]);
  float x1 = zb[t + 256] + gamma * (acc1 + b_base[t + 256]);

  red[t] = x0 + x1;
  __syncthreads();
  for (int off = 128; off > 0; off >>= 1) {
    if (t < off) red[t] += red[t + off];
    __syncthreads();
  }
  const float mean = red[0] / (float)DMODEL;
  __syncthreads();
  red[t] = (x0 - mean) * (x0 - mean) + (x1 - mean) * (x1 - mean);
  __syncthreads();
  for (int off = 128; off > 0; off >>= 1) {
    if (t < off) red[t] += red[t + off];
    __syncthreads();
  }
  const float var = red[0] / (float)DMODEL;
  const float inv = rsqrtf(var + 1e-5f);
  out[(size_t)b * DMODEL + t]       = (x0 - mean) * inv * ln_scale[t] + ln_bias[t];
  out[(size_t)b * DMODEL + t + 256] = (x1 - mean) * inv * ln_scale[t + 256] + ln_bias[t + 256];
}

// ---------------------------------------------------------------------------
// Launch
// ---------------------------------------------------------------------------
extern "C" void kernel_launch(void* const* d_in, const int* in_sizes, int n_in,
                              void* d_out, int out_size, void* d_ws, size_t ws_size,
                              hipStream_t stream) {
  const float* z       = (const float*)d_in[0];
  const float* pool    = (const float*)d_in[1];
  const float* W_Q     = (const float*)d_in[2];
  const float* W_K     = (const float*)d_in[3];
  const float* logits  = (const float*)d_in[4];
  const float* tau     = (const float*)d_in[5];
  const float* W_base  = (const float*)d_in[6];
  const float* b_base  = (const float*)d_in[7];
  const float* gamma   = (const float*)d_in[8];
  const float* ln_s    = (const float*)d_in[9];
  const float* ln_b    = (const float*)d_in[10];
  float* out = (float*)d_out;

  // workspace carve-up (256B aligned), ~12.5 MB total
  char* p = (char*)d_ws;
  auto alloc = [&](size_t bytes) {
    char* r = p;
    p += (bytes + 255) & ~(size_t)255;
    return r;
  };
  __bf16* wkt  = (__bf16*)alloc((size_t)QKDIM * DPOOL  * 2);  // 4 MB
  __bf16* wqt  = (__bf16*)alloc((size_t)QKDIM * DMODEL * 2);  // 128 KB
  __bf16* wbb  = (__bf16*)alloc((size_t)DMODEL * DMODEL * 2); // 512 KB
  float*  qf   = (float*) alloc((size_t)BB * QKDIM * 4);      // 512 KB
  float*  kf   = (float*) alloc((size_t)NPOOL * QKDIM * 4);   // 512 KB
  __bf16* qn   = (__bf16*)alloc((size_t)BB * QKDIM * 2);      // 256 KB
  __bf16* kn   = (__bf16*)alloc((size_t)NPOOL * QKDIM * 2);   // 256 KB
  float*  sc   = (float*) alloc((size_t)BB * NPOOL * 4);      // 4 MB
  float*  hb   = (float*) alloc((size_t)BB * DMODEL * 4);     // 2 MB
  float*  alp  = (float*) alloc((size_t)BB * KMAX * 4);       // 64 KB
  int*    idx  = (int*)   alloc((size_t)BB * KMAX * 4);       // 64 KB

  // 1. operand prep: transpose+convert W_K, W_Q; convert W_base to bf16
  {
    int totK = SS * DPOOL * DKQ;   // 2,097,152
    transpose_w_bf16<<<(totK + 255) / 256, 256, 0, stream>>>(W_K, wkt, DPOOL, totK);
    int totQ = SS * DMODEL * DKQ;  // 65,536
    transpose_w_bf16<<<(totQ + 255) / 256, 256, 0, stream>>>(W_Q, wqt, DMODEL, totQ);
    int totB = DMODEL * DMODEL;    // 262,144
    convert_bf16<<<(totB + 255) / 256, 256, 0, stream>>>(W_base, wbb, totB);
  }

  // 2. keys = pool @ W_K: TDM-staged A, streams 64 MB of pool once
  wmma_gemm_pool_tdm<<<dim3(QKDIM / 64, NPOOL / 64), 128, 0, stream>>>(pool, wkt, kf);

  // 3. queries = z @ W_Q
  wmma_gemm_nt<float><<<dim3(QKDIM / 64, BB / 64), 128, 0, stream>>>(
      z, wqt, qf, BB, QKDIM, DMODEL, DMODEL, DMODEL, QKDIM);

  // 4. normalize per aspect, fold sqrt(softmax(logits))
  normalize_qk<<<BB, QKDIM, 0, stream>>>(qf, qn, logits);
  normalize_qk<<<NPOOL, QKDIM, 0, stream>>>(kf, kn, logits);

  // 5. scores = qn @ kn^T  (L2-resident bf16 operands)
  wmma_gemm_nt<__bf16><<<dim3(NPOOL / 64, BB / 64), 128, 0, stream>>>(
      qn, kn, sc, BB, NPOOL, QKDIM, QKDIM, QKDIM, NPOOL);

  // 6. top-16 + sparse alpha
  topk_alpha<<<BB, 32, 0, stream>>>(sc, tau, alp, idx);

  // 7. h_base = z @ W_base^T
  wmma_gemm_nt<float><<<dim3(DMODEL / 64, BB / 64), 128, 0, stream>>>(
      z, wbb, hb, BB, DMODEL, DMODEL, DMODEL, DMODEL, DMODEL);

  // 8. sparse rank-8 delta + bias + residual + LayerNorm (never builds W_asm)
  assemble_ln<<<BB, 256, 0, stream>>>(z, pool, hb, alp, idx, b_base, gamma, ln_s, ln_b, out);
}
